// BlockRecurrentTransformer_74775380623445
// MI455X (gfx1250) — matrix operand order repaired and checked by reading, not compiled, not run on hardware
//
#include <hip/hip_runtime.h>
#include <cstdint>
#include <cstddef>

// ---------------------------------------------------------------------------
// Types / constants
// ---------------------------------------------------------------------------
typedef __attribute__((ext_vector_type(16))) __bf16 v16bf;
typedef __attribute__((ext_vector_type(8)))  __bf16 v8bf;
typedef __attribute__((ext_vector_type(8)))  float  v8f;

#define Hh     16
#define Dd     64
#define Nn     4096
#define DIMd   1024
#define WIDTHw 512
#define NSs    512
#define NPAD   (Nn + WIDTHw)     /* 4608: zero block prepended for block attn */

#define WMMA_BF16(a, b, c) \
  __builtin_amdgcn_wmma_f32_16x16x32_bf16(false, (a), false, (b), (short)0, (c), false, false)

// ---------------------------------------------------------------------------
// WMMA fragment loaders — all paths use 128-bit loads.
// A 16x32 bf16 (row-major, ld): lane m=lane&15, kh=(lane>=16)?8:0;
//   values = rows[kh..kh+7] ++ rows[kh+16..kh+23]  -> two b128 loads.
// ---------------------------------------------------------------------------
static __device__ __forceinline__ v16bf frag_a(const __bf16* p, int ld) {
  int lane = threadIdx.x & 31;
  const __bf16* r = p + (size_t)(lane & 15) * ld + ((lane >> 4) ? 8 : 0);
  v8bf lo = *(const v8bf*)r;
  v8bf hi = *(const v8bf*)(r + 16);
  return __builtin_shufflevector(lo, hi, 0, 1, 2, 3, 4, 5, 6, 7,
                                 8, 9, 10, 11, 12, 13, 14, 15);
}

// B 32x16 bf16 from a TRANSPOSED [n][k] buffer (ld = k-pitch):
//   lane n=lane&15, kh=(lane>=16)?8:0; values = row n, k=kh..kh+7 ++ kh+16..kh+23
//   -> two b128 loads.
static __device__ __forceinline__ v16bf frag_b_tr(const __bf16* p, int ld) {
  int lane = threadIdx.x & 31;
  const __bf16* r = p + (size_t)(lane & 15) * ld + ((lane >> 4) ? 8 : 0);
  v8bf lo = *(const v8bf*)r;
  v8bf hi = *(const v8bf*)(r + 16);
  return __builtin_shufflevector(lo, hi, 0, 1, 2, 3, 4, 5, 6, 7,
                                 8, 9, 10, 11, 12, 13, 14, 15);
}

// ---------------------------------------------------------------------------
// GEMM: C[M,N](f32) = A[M,K](bf16, row major) @ B[K,N](bf16, row major)
// 64x64 block tile, 8 waves; B staged transposed in LDS so fragment reads
// are ds_load_b128 pairs.
// ---------------------------------------------------------------------------
#define BT_LD 40   /* transposed-B pitch: 40*2B = 80B keeps 16B alignment, kills conflicts */

__global__ __launch_bounds__(256) void wmma_gemm_bf16(
    const __bf16* __restrict__ A, const __bf16* __restrict__ B,
    float* __restrict__ C, int M, int N, int K) {
  __shared__ __attribute__((aligned(16))) __bf16 sA[64 * 32];
  __shared__ __attribute__((aligned(16))) __bf16 sBt[64 * BT_LD];

  int tid = threadIdx.x;
  int m0  = blockIdx.y * 64;
  int n0  = blockIdx.x * 64;
  int wav = tid >> 5;
  int wr  = wav >> 1;        // 0..3 row tile
  int wcp = wav & 1;         // 0..1 col pair
  int lane  = tid & 31;
  int nl    = lane & 15;
  int rbase = (lane >> 4) * 8;

  v8f acc0 = {};
  v8f acc1 = {};

  int arow = tid >> 2;             // 0..63
  int acol = (tid & 3) * 8;        // 0,8,16,24
  int brow = tid >> 3;             // 0..31  (k)
  int bcol = (tid & 7) * 8;        // 0..56  (n)

  for (int kb = 0; kb < K; kb += 32) {
    const __bf16* ga = A + (size_t)(m0 + arow) * K + kb + acol;
    const __bf16* gb = B + (size_t)(kb + brow) * N + n0 + bcol;
    *(uint4*)(&sA[arow * 32 + acol]) = *(const uint4*)ga;
    v8bf bv = *(const v8bf*)gb;
#pragma unroll
    for (int j = 0; j < 8; ++j)           // transpose into sBt[n][k]
      sBt[(bcol + j) * BT_LD + brow] = bv[j];
    if (kb + 32 < K) {                    // gfx1250 global_prefetch_b8
      __builtin_prefetch(ga + 32, 0, 1);
      __builtin_prefetch(B + (size_t)(kb + 32 + brow) * N + n0 + bcol, 0, 1);
    }
    __syncthreads();

    v16bf a  = frag_a(&sA[wr * 16 * 32], 32);
    v16bf b0 = frag_b_tr(&sBt[(wcp * 32) * BT_LD], BT_LD);
    v16bf b1 = frag_b_tr(&sBt[(wcp * 32 + 16) * BT_LD], BT_LD);
    acc0 = WMMA_BF16(a, b0, acc0);
    acc1 = WMMA_BF16(a, b1, acc1);
    __syncthreads();
  }

#pragma unroll
  for (int r = 0; r < 8; ++r) {
    int row = m0 + wr * 16 + rbase + r;
    C[(size_t)row * N + n0 + wcp * 32 + nl]      = acc0[r];
    C[(size_t)row * N + n0 + wcp * 32 + 16 + nl] = acc1[r];
  }
}

// ---------------------------------------------------------------------------
// Flash attention (online softmax), generic across the four attentions.
// Q : [H][nq_tot][64] bf16 (pre l2norm * q_scale * 8)
// K : [H][nk_tot][64] bf16 (token-major; pre l2norm * k_scale)
// VT: [H][64][nk_tot] bf16 (dim-major)
// One workgroup = one (window z, head, 64-query tile). 256 threads / 8 waves.
// Per key chunk: stage sK[key][d] + sVt[d][key] with coalesced b128 copies.
// ---------------------------------------------------------------------------
__global__ __launch_bounds__(256) void flash_attn(
    const __bf16* __restrict__ Q, const __bf16* __restrict__ K,
    const __bf16* __restrict__ VT, const float* __restrict__ bias,
    float* __restrict__ Out, int out_ld, int out_col,
    int nq_tot, int nk_tot, int window_q, int key_stride, int key_len,
    int causal, int causal_off) {
  __shared__ float  sS[64 * 64];
  __shared__ __attribute__((aligned(16))) __bf16 sP[64 * 64];
  __shared__ __attribute__((aligned(16))) __bf16 sK[64 * 64];   // [key][d]
  __shared__ __attribute__((aligned(16))) __bf16 sVt[64 * 64];  // [d][key]
  __shared__ float  s_alpha[64];
  __shared__ float  s_l[64];

  int tid  = threadIdx.x;
  int h    = blockIdx.y;
  int z    = blockIdx.z;
  int bx64 = blockIdx.x * 64;            // query tile base within window
  int qbase = z * window_q + bx64;       // global query row
  int key0  = z * key_stride;            // base into (padded) key space

  int wav   = tid >> 5;
  int lane  = tid & 31;
  int nl    = lane & 15;
  int rbase = (lane >> 4) * 8;
  int srow  = wav >> 1;                  // 0..3
  int scp   = wav & 1;                   // 0..1

  // staging coords
  int r0 = tid >> 2;                     // 0..63
  int c0 = (tid & 3) * 16;               // 0,16,32,48

  // Q fragments (rows srow*16..+15, K-dim = 64 -> two 32-wide frags)
  const __bf16* qp = Q + ((size_t)h * nq_tot + qbase + srow * 16) * Dd;
  v16bf aq0 = frag_a(qp, Dd);
  v16bf aq1 = frag_a(qp + 32, Dd);

  v8f o0 = {};
  v8f o1 = {};
  float m_run = -3.4e38f, l_run = 0.f;

  int nchunk = key_len / 64;
  for (int c = 0; c < nchunk; ++c) {
    if (causal && (c * 64 > bx64 + 63 + causal_off)) break;  // uniform skip

    // ---- stage K/V chunk into LDS (coalesced b128) ----
    {
      const __bf16* gk = K + ((size_t)h * nk_tot + key0 + c * 64 + r0) * Dd + c0;
      *(uint4*)&sK[r0 * 64 + c0]     = *(const uint4*)gk;
      *(uint4*)&sK[r0 * 64 + c0 + 8] = *(const uint4*)(gk + 8);
      const __bf16* gv = VT + ((size_t)h * Dd + r0) * nk_tot + key0 + c * 64 + c0;
      *(uint4*)&sVt[r0 * 64 + c0]     = *(const uint4*)gv;
      *(uint4*)&sVt[r0 * 64 + c0 + 8] = *(const uint4*)(gv + 8);
    }
    __syncthreads();

    // ---- S chunk: two 16x16 score tiles per wave, bias folded into C ----
    for (int t = 0; t < 2; ++t) {
      int sct = scp * 2 + t;
      v8f acc;
#pragma unroll
      for (int r = 0; r < 8; ++r) {
        int i_loc = bx64 + srow * 16 + rbase + r;
        int j_win = c * 64 + sct * 16 + nl;
        acc[r] = bias ? bias[((size_t)h * WIDTHw + i_loc) * (2 * WIDTHw) + j_win] : 0.f;
      }
      v16bf b0 = frag_b_tr(&sK[(sct * 16) * 64], 64);       // d 0..31
      v16bf b1 = frag_b_tr(&sK[(sct * 16) * 64 + 32], 64);  // d 32..63
      acc = WMMA_BF16(aq0, b0, acc);
      acc = WMMA_BF16(aq1, b1, acc);
#pragma unroll
      for (int r = 0; r < 8; ++r) {
        int i_loc = bx64 + srow * 16 + rbase + r;
        int j_win = c * 64 + sct * 16 + nl;
        float s = acc[r];
        if (causal && (j_win > i_loc + causal_off)) s = -1e30f;
        sS[(srow * 16 + rbase + r) * 64 + sct * 16 + nl] = s;
      }
    }
    __syncthreads();

    // ---- online softmax, one row per thread (tid < 64) ----
    if (tid < 64) {
      const float* sr = &sS[tid * 64];
      float mc = -3.4e38f;
      for (int j = 0; j < 64; ++j) mc = fmaxf(mc, sr[j]);
      float mnew  = fmaxf(m_run, mc);
      float alpha = (m_run <= -3.0e38f) ? 0.f : __expf(m_run - mnew);
      float lc = 0.f;
      if (mnew <= -1e29f) {
        for (int j = 0; j < 64; ++j) sP[tid * 64 + j] = (__bf16)0.f;
      } else {
        for (int j = 0; j < 64; ++j) {
          float p = __expf(sr[j] - mnew);
          sP[tid * 64 + j] = (__bf16)p;
          lc += p;
        }
      }
      l_run = l_run * alpha + lc;
      m_run = mnew;
      s_alpha[tid] = alpha;
    }
    __syncthreads();

    // ---- O = O*alpha + P @ V_chunk ----
#pragma unroll
    for (int r = 0; r < 8; ++r) {
      float a = s_alpha[srow * 16 + rbase + r];
      o0[r] *= a;
      o1[r] *= a;
    }
    v16bf p0 = frag_a(&sP[srow * 16 * 64], 64);        // keys 0..31
    v16bf p1 = frag_a(&sP[srow * 16 * 64 + 32], 64);   // keys 32..63
    v16bf vf0 = frag_b_tr(&sVt[(scp * 2) * 16 * 64], 64);        // keys 0..31
    v16bf vf1 = frag_b_tr(&sVt[(scp * 2) * 16 * 64 + 32], 64);   // keys 32..63
    o0 = WMMA_BF16(p0, vf0, o0);
    o0 = WMMA_BF16(p1, vf1, o0);
    v16bf vg0 = frag_b_tr(&sVt[(scp * 2 + 1) * 16 * 64], 64);
    v16bf vg1 = frag_b_tr(&sVt[(scp * 2 + 1) * 16 * 64 + 32], 64);
    o1 = WMMA_BF16(p0, vg0, o1);
    o1 = WMMA_BF16(p1, vg1, o1);
    __syncthreads();
  }

  if (tid < 64) s_l[tid] = l_run;
  __syncthreads();

#pragma unroll
  for (int r = 0; r < 8; ++r) {
    int irow = srow * 16 + rbase + r;
    float lv = s_l[irow];
    float inv = (lv > 0.f) ? (1.f / lv) : 0.f;
    float* orow = Out + (size_t)(qbase + irow) * out_ld + out_col + h * Dd;
    orow[(scp * 2) * 16 + nl]     = o0[r] * inv;
    orow[(scp * 2 + 1) * 16 + nl] = o1[r] * inv;
  }
}

// ---------------------------------------------------------------------------
// LayerNorm (+ optional additive term), f32 -> bf16
// ---------------------------------------------------------------------------
__global__ __launch_bounds__(256) void ln_bf16_kernel(
    const float* __restrict__ x, const float* __restrict__ g,
    const float* __restrict__ add, __bf16* __restrict__ out, int dim) {
  __shared__ float rs[256], rs2[256];
  int row = blockIdx.x, tid = threadIdx.x;
  const float* xr = x + (size_t)row * dim;
  float s = 0.f, s2 = 0.f;
  for (int i = tid; i < dim; i += 256) {
    float v = xr[i];
    s += v;
    s2 += v * v;
  }
  rs[tid] = s; rs2[tid] = s2;
  __syncthreads();
  for (int st = 128; st; st >>= 1) {
    if (tid < st) { rs[tid] += rs[tid + st]; rs2[tid] += rs2[tid + st]; }
    __syncthreads();
  }
  float mu  = rs[0] / dim;
  float var = rs2[0] / dim - mu * mu;
  float inv = rsqrtf(var + 1e-5f);
  for (int i = tid; i < dim; i += 256) {
    float v = (xr[i] - mu) * inv * g[i];
    if (add) v += add[(size_t)row * dim + i];
    out[(size_t)row * dim + i] = (__bf16)v;
  }
}

// ---------------------------------------------------------------------------
// Elementwise helpers
// ---------------------------------------------------------------------------
__global__ void f32_to_bf16_kernel(const float* __restrict__ src,
                                   __bf16* __restrict__ dst, size_t n) {
  for (size_t i = (size_t)blockIdx.x * blockDim.x + threadIdx.x; i < n;
       i += (size_t)gridDim.x * blockDim.x)
    dst[i] = (__bf16)src[i];
}

__global__ void zero_bf16_kernel(__bf16* __restrict__ p, size_t n) {
  for (size_t i = (size_t)blockIdx.x * blockDim.x + threadIdx.x; i < n;
       i += (size_t)gridDim.x * blockDim.x)
    p[i] = (__bf16)0.f;
}

// qk-rmsnorm + head repack. mode 0: Q [H][ntok][64]; 1: dim-major [H][64][nk_pad];
// 2: token-major [H][nk_pad][64]. grid = (ntok, H), 64 threads.
__global__ __launch_bounds__(64) void pack_heads_kernel(
    const float* __restrict__ src, int src_ld, int src_off,
    const float* __restrict__ scale, float mul, int do_norm,
    __bf16* __restrict__ dst, int ntok, int nk_pad, int pad0, int mode) {
  __shared__ float red[64];
  int t = blockIdx.x, h = blockIdx.y, d = threadIdx.x;
  float v = src[(size_t)t * src_ld + src_off + h * Dd + d];
  if (do_norm) {
    red[d] = v * v;
    __syncthreads();
    for (int s = 32; s; s >>= 1) {
      if (d < s) red[d] += red[d + s];
      __syncthreads();
    }
    float den = fmaxf(sqrtf(red[0]), 1e-12f);
    v = v / den * scale[d] * mul;
  }
  size_t idx;
  if (mode == 0)      idx = ((size_t)h * ntok + t) * Dd + d;
  else if (mode == 1) idx = ((size_t)h * Dd + d) * nk_pad + pad0 + t;
  else                idx = ((size_t)h * nk_pad + pad0 + t) * Dd + d;
  dst[idx] = (__bf16)v;
}

// memories output: [2][H][512][64] from raw k/v of the last block
__global__ void memories_kernel(const float* __restrict__ qkv,
                                float* __restrict__ mem) {
  size_t total = (size_t)2 * Hh * WIDTHw * Dd;
  for (size_t idx = (size_t)blockIdx.x * blockDim.x + threadIdx.x; idx < total;
       idx += (size_t)gridDim.x * blockDim.x) {
    size_t half = idx / ((size_t)Hh * WIDTHw * Dd);
    size_t rem  = idx % ((size_t)Hh * WIDTHw * Dd);
    int h = (int)(rem / (WIDTHw * Dd));
    int i = (int)((rem / Dd) % WIDTHw);
    int d = (int)(rem % Dd);
    int col = (half ? 2048 : 1024) + h * Dd + d;
    mem[idx] = qkv[(size_t)(Nn - WIDTHw + i) * (3 * DIMd) + col];
  }
}

// new_states = sigmoid(beta)*(zlin + b_gate) + (1-sigmoid(beta))*init_state
__global__ void gate_kernel(const float* __restrict__ zlin,
                            const float* __restrict__ b_gate,
                            const float* __restrict__ beta,
                            const float* __restrict__ init_state,
                            float* __restrict__ out) {
  size_t total = (size_t)NSs * DIMd;
  for (size_t idx = (size_t)blockIdx.x * blockDim.x + threadIdx.x; idx < total;
       idx += (size_t)gridDim.x * blockDim.x) {
    int dim = (int)(idx % DIMd);
    float sg = 1.f / (1.f + __expf(-beta[dim]));
    out[idx] = sg * (zlin[idx] + b_gate[dim]) + (1.f - sg) * init_state[idx];
  }
}

// ---------------------------------------------------------------------------
// Orchestration
// ---------------------------------------------------------------------------
extern "C" void kernel_launch(void* const* d_in, const int* in_sizes, int n_in,
                              void* d_out, int out_size, void* d_ws, size_t ws_size,
                              hipStream_t stream) {
  (void)in_sizes; (void)n_in; (void)out_size; (void)ws_size;
  const float* x        = (const float*)d_in[0];
  const float* rel_bias = (const float*)d_in[1];
  const float* gamma    = (const float*)d_in[2];
  const float* w_qkv    = (const float*)d_in[3];
  const float* q_scale  = (const float*)d_in[4];
  const float* k_scale  = (const float*)d_in[5];
  const float* w_out    = (const float*)d_in[6];
  const float* sgamma   = (const float*)d_in[7];
  const float* w_q2s    = (const float*)d_in[8];
  const float* w_qfs    = (const float*)d_in[9];
  const float* w_sqkv   = (const float*)d_in[10];
  const float* init_st  = (const float*)d_in[11];
  const float* pos_ids  = (const float*)d_in[12];
  const float* w_sout   = (const float*)d_in[13];
  const float* ts_q     = (const float*)d_in[14];
  const float* ts_k     = (const float*)d_in[15];
  const float* ss_q     = (const float*)d_in[16];
  const float* ss_k     = (const float*)d_in[17];
  const float* fs_q     = (const float*)d_in[18];
  const float* fs_k     = (const float*)d_in[19];
  const float* w_gate   = (const float*)d_in[20];
  const float* b_gate   = (const float*)d_in[21];
  const float* beta     = (const float*)d_in[22];

  float* out_final = (float*)d_out;                         // [4096,1024]
  float* out_mem   = out_final + (size_t)Nn * DIMd;         // [2,16,512,64]
  float* out_newst = out_mem + (size_t)2 * Hh * WIDTHw * Dd;// [512,1024]

  // ---- workspace bump allocator ----
  size_t off = 0;
  auto alloc = [&](size_t bytes) -> char* {
    char* p = (char*)d_ws + off;
    off = (off + bytes + 255) & ~(size_t)255;
    return p;
  };
  __bf16* xn_bf      = (__bf16*)alloc((size_t)Nn * DIMd * 2);
  __bf16* w_qkv_bf   = (__bf16*)alloc((size_t)DIMd * 3 * DIMd * 2);
  __bf16* w_q2s_bf   = (__bf16*)alloc((size_t)DIMd * DIMd * 2);
  __bf16* w_qfs_bf   = (__bf16*)alloc((size_t)DIMd * DIMd * 2);
  __bf16* w_sqkv_bf  = (__bf16*)alloc((size_t)DIMd * 3 * DIMd * 2);
  __bf16* w_out_bf   = (__bf16*)alloc((size_t)2 * DIMd * DIMd * 2);
  __bf16* w_sout_bf  = (__bf16*)alloc((size_t)2 * DIMd * DIMd * 2);
  __bf16* w_gate_bf  = (__bf16*)alloc((size_t)DIMd * DIMd * 2);
  float*  qkv        = (float*)alloc((size_t)Nn * 3 * DIMd * 4);
  float*  q2s        = (float*)alloc((size_t)Nn * DIMd * 4);
  __bf16* st_bf      = (__bf16*)alloc((size_t)NSs * DIMd * 2);
  __bf16* init_bf    = (__bf16*)alloc((size_t)NSs * DIMd * 2);
  float*  qfs        = (float*)alloc((size_t)NSs * DIMd * 4);
  float*  sqkv       = (float*)alloc((size_t)NSs * 3 * DIMd * 4);
  __bf16* Qb         = (__bf16*)alloc((size_t)Hh * Nn * Dd * 2);
  __bf16* Kb         = (__bf16*)alloc((size_t)Hh * NPAD * Dd * 2);   // token-major
  __bf16* VTb        = (__bf16*)alloc((size_t)Hh * Dd * NPAD * 2);   // dim-major
  __bf16* Q2S_bf     = (__bf16*)alloc((size_t)Hh * Nn * Dd * 2);
  __bf16* Kts        = (__bf16*)alloc((size_t)Hh * NSs * Dd * 2);
  __bf16* VTst       = (__bf16*)alloc((size_t)Hh * Dd * NSs * 2);
  __bf16* Qss        = (__bf16*)alloc((size_t)Hh * NSs * Dd * 2);
  __bf16* Kss        = (__bf16*)alloc((size_t)Hh * NSs * Dd * 2);
  __bf16* Qfs_bf     = (__bf16*)alloc((size_t)Hh * NSs * Dd * 2);
  __bf16* Kfs        = (__bf16*)alloc((size_t)Hh * NSs * Dd * 2);
  __bf16* VTfs       = (__bf16*)alloc((size_t)Hh * Dd * NSs * 2);
  float*  out_cat    = (float*)alloc((size_t)Nn * 2 * DIMd * 4);
  __bf16* out_cat_bf = (__bf16*)alloc((size_t)Nn * 2 * DIMd * 2);
  float*  st_cat     = (float*)alloc((size_t)NSs * 2 * DIMd * 4);
  __bf16* st_cat_bf  = (__bf16*)alloc((size_t)NSs * 2 * DIMd * 2);
  float*  state_out  = (float*)alloc((size_t)NSs * DIMd * 4);
  __bf16* state_o_bf = (__bf16*)alloc((size_t)NSs * DIMd * 2);
  float*  zlin       = (float*)alloc((size_t)NSs * DIMd * 4);

  auto cvt = [&](const float* s, __bf16* d, size_t n) {
    f32_to_bf16_kernel<<<2048, 256, 0, stream>>>(s, d, n);
  };

  // 1. LayerNorms + weight converts
  ln_bf16_kernel<<<Nn, 256, 0, stream>>>(x, gamma, nullptr, xn_bf, DIMd);
  ln_bf16_kernel<<<NSs, 256, 0, stream>>>(init_st, sgamma, pos_ids, st_bf, DIMd);
  cvt(w_qkv,  w_qkv_bf,  (size_t)DIMd * 3 * DIMd);
  cvt(w_q2s,  w_q2s_bf,  (size_t)DIMd * DIMd);
  cvt(w_qfs,  w_qfs_bf,  (size_t)DIMd * DIMd);
  cvt(w_sqkv, w_sqkv_bf, (size_t)DIMd * 3 * DIMd);
  cvt(w_out,  w_out_bf,  (size_t)2 * DIMd * DIMd);
  cvt(w_sout, w_sout_bf, (size_t)2 * DIMd * DIMd);
  cvt(w_gate, w_gate_bf, (size_t)DIMd * DIMd);
  cvt(init_st, init_bf,  (size_t)NSs * DIMd);

  // 2. Projection GEMMs
  wmma_gemm_bf16<<<dim3(48, 64), 256, 0, stream>>>(xn_bf, w_qkv_bf, qkv, Nn, 3 * DIMd, DIMd);
  wmma_gemm_bf16<<<dim3(16, 64), 256, 0, stream>>>(xn_bf, w_q2s_bf, q2s, Nn, DIMd, DIMd);
  wmma_gemm_bf16<<<dim3(16, 8), 256, 0, stream>>>(st_bf, w_qfs_bf, qfs, NSs, DIMd, DIMd);
  wmma_gemm_bf16<<<dim3(48, 8), 256, 0, stream>>>(init_bf, w_sqkv_bf, sqkv, NSs, 3 * DIMd, DIMd);

  // 3. Head packing (+ zero block for block-attention K/V)
  zero_bf16_kernel<<<2048, 256, 0, stream>>>(Kb, (size_t)Hh * NPAD * Dd);
  zero_bf16_kernel<<<2048, 256, 0, stream>>>(VTb, (size_t)Hh * Dd * NPAD);
  pack_heads_kernel<<<dim3(Nn, Hh), 64, 0, stream>>>(qkv, 3 * DIMd, 0, q_scale, 8.f, 1, Qb, Nn, 0, 0, 0);
  pack_heads_kernel<<<dim3(Nn, Hh), 64, 0, stream>>>(qkv, 3 * DIMd, DIMd, k_scale, 1.f, 1, Kb, Nn, NPAD, WIDTHw, 2);
  pack_heads_kernel<<<dim3(Nn, Hh), 64, 0, stream>>>(qkv, 3 * DIMd, 2 * DIMd, nullptr, 1.f, 0, VTb, Nn, NPAD, WIDTHw, 1);
  pack_heads_kernel<<<dim3(Nn, Hh), 64, 0, stream>>>(q2s, DIMd, 0, ts_q, 8.f, 1, Q2S_bf, Nn, 0, 0, 0);
  pack_heads_kernel<<<dim3(NSs, Hh), 64, 0, stream>>>(sqkv, 3 * DIMd, DIMd, ts_k, 1.f, 1, Kts, NSs, NSs, 0, 2);
  pack_heads_kernel<<<dim3(NSs, Hh), 64, 0, stream>>>(sqkv, 3 * DIMd, 2 * DIMd, nullptr, 1.f, 0, VTst, NSs, NSs, 0, 1);
  pack_heads_kernel<<<dim3(NSs, Hh), 64, 0, stream>>>(sqkv, 3 * DIMd, 0, ss_q, 8.f, 1, Qss, NSs, 0, 0, 0);
  pack_heads_kernel<<<dim3(NSs, Hh), 64, 0, stream>>>(sqkv, 3 * DIMd, DIMd, ss_k, 1.f, 1, Kss, NSs, NSs, 0, 2);
  pack_heads_kernel<<<dim3(NSs, Hh), 64, 0, stream>>>(qfs, DIMd, 0, fs_q, 8.f, 1, Qfs_bf, NSs, 0, 0, 0);
  pack_heads_kernel<<<dim3(NSs, Hh), 64, 0, stream>>>(qkv + (size_t)(Nn - WIDTHw) * 3 * DIMd, 3 * DIMd, DIMd, fs_k, 1.f, 1, Kfs, NSs, NSs, 0, 2);
  pack_heads_kernel<<<dim3(NSs, Hh), 64, 0, stream>>>(qkv + (size_t)(Nn - WIDTHw) * 3 * DIMd, 3 * DIMd, 2 * DIMd, nullptr, 1.f, 0, VTfs, NSs, NSs, 0, 1);

  // 4. Attentions
  // block attention: 8 windows of 512 queries, 1024-key window, causal+bias
  flash_attn<<<dim3(8, Hh, 8), 256, 0, stream>>>(Qb, Kb, VTb, rel_bias,
      out_cat, 2 * DIMd, 0, Nn, NPAD, WIDTHw, WIDTHw, 2 * WIDTHw, 1, WIDTHw);
  // to-state attention: all 4096 queries vs 512 state keys
  flash_attn<<<dim3(64, Hh, 1), 256, 0, stream>>>(Q2S_bf, Kts, VTst, nullptr,
      out_cat, 2 * DIMd, DIMd, Nn, NSs, Nn, 0, NSs, 0, 0);
  // state self-attention
  flash_attn<<<dim3(8, Hh, 1), 256, 0, stream>>>(Qss, Kss, VTst, nullptr,
      st_cat, 2 * DIMd, 0, NSs, NSs, NSs, 0, NSs, 0, 0);
  // from-state attention (vs last-block memories)
  flash_attn<<<dim3(8, Hh, 1), 256, 0, stream>>>(Qfs_bf, Kfs, VTfs, nullptr,
      st_cat, 2 * DIMd, DIMd, NSs, NSs, NSs, 0, NSs, 0, 0);

  // 5. Output projections
  cvt(out_cat, out_cat_bf, (size_t)Nn * 2 * DIMd);
  wmma_gemm_bf16<<<dim3(16, 64), 256, 0, stream>>>(out_cat_bf, w_out_bf, out_final, Nn, DIMd, 2 * DIMd);
  cvt(st_cat, st_cat_bf, (size_t)NSs * 2 * DIMd);
  wmma_gemm_bf16<<<dim3(16, 8), 256, 0, stream>>>(st_cat_bf, w_sout_bf, state_out, NSs, DIMd, 2 * DIMd);
  cvt(state_out, state_o_bf, (size_t)NSs * DIMd);
  wmma_gemm_bf16<<<dim3(16, 8), 256, 0, stream>>>(state_o_bf, w_gate_bf, zlin, NSs, DIMd, DIMd);

  // 6. Epilogues: memories copy + sigmoid-EMA gate
  memories_kernel<<<1024, 256, 0, stream>>>(qkv, out_mem);
  gate_kernel<<<512, 256, 0, stream>>>(zlin, b_gate, beta, init_st, out_newst);
}
